// SpatialCrossAttention_17265768530479
// MI455X (gfx1250) — matrix operand order repaired and checked by reading, not compiled, not run on hardware
//
#include <hip/hip_runtime.h>

#define NQ      22500
#define EMBED   256
#define HEADS   8
#define HEAD_DIM 32
#define LEVELS  4
#define POINTS  4
#define CAMS    6
#define TOTHW   30825   // 23200 + 5800 + 1450 + 375

typedef __attribute__((ext_vector_type(16))) _Float16 v16h;
typedef __attribute__((ext_vector_type(2)))  _Float16 h2;
typedef __attribute__((ext_vector_type(8)))  float    v8f;

#define LDSA_STRIDE 34   // 32 halves + pad (keeps 4B alignment, breaks bank conflicts)
#define LDSB_STRIDE 34

// ---------------------------------------------------------------------------
// WMMA fragment loaders (CDNA5 16-bit layouts, CDNA5 ISA §7.12.2)
// A 16x32 f16: lanes 0-15 -> M=lane, K pairs {0..7,16..23}; lanes 16-31 +8.
// B 32x16 f16: lanes 0-15 -> N=lane, K pairs {0..7,16..23}; lanes 16-31 +8.
// ---------------------------------------------------------------------------
__device__ inline v16h load_frag_a(const _Float16* As, int m0) {
    int lane = threadIdx.x & 31;
    int m  = m0 + (lane & 15);
    int hi = (lane >> 4) << 3;            // 0 or 8
    const _Float16* row = As + m * LDSA_STRIDE;
    v16h a;
#pragma unroll
    for (int j = 0; j < 8; ++j) {
        int k = 2 * j + ((j >= 4) ? 8 : 0) + hi;
        h2 pr = *(const h2*)(row + k);
        a[2 * j]     = pr.x;
        a[2 * j + 1] = pr.y;
    }
    return a;
}

// BsT is stored transposed: BsT[n * LDSB_STRIDE + k]
__device__ inline v16h load_frag_b(const _Float16* BsT, int n0) {
    int lane = threadIdx.x & 31;
    int n  = n0 + (lane & 15);
    int hi = (lane >> 4) << 3;            // 0 or 8
    const _Float16* col = BsT + n * LDSB_STRIDE;
    v16h b;
#pragma unroll
    for (int j = 0; j < 8; ++j) {
        h2 pr = *(const h2*)(col + 2 * j + ((j >= 4) ? 8 : 0) + hi);
        b[2 * j]     = pr.x;
        b[2 * j + 1] = pr.y;
    }
    return b;
}

#define WMMA_F16(A, B, C) \
    __builtin_amdgcn_wmma_f32_16x16x32_f16(false, (A), false, (B), (short)0, (C), false, false)

// ---------------------------------------------------------------------------
// Kernel 1: per-(cam,query) point projection -> ref2d, mask
// ---------------------------------------------------------------------------
__global__ __launch_bounds__(256) void point_kernel(
    const float* __restrict__ l2i, const float* __restrict__ refp,
    const int* __restrict__ img_h_p, const int* __restrict__ img_w_p,
    float* __restrict__ ref2d, float* __restrict__ mask)
{
    int i = blockIdx.x * blockDim.x + threadIdx.x;
    if (i >= CAMS * NQ) return;
    int cam = i / NQ, q = i - cam * NQ;
    const float* M = l2i + cam * 16;
    float x = refp[q * 3 + 0], y = refp[q * 3 + 1], z = refp[q * 3 + 2];
    float cx = M[0] * x + M[1] * y + M[2]  * z + M[3];
    float cy = M[4] * x + M[5] * y + M[6]  * z + M[7];
    float cz = M[8] * x + M[9] * y + M[10] * z + M[11];
    const float eps = 1e-5f;
    float inv = 1.0f / (cz + eps);
    float un = (cx * inv) / (float)img_w_p[0];
    float vn = (cy * inv) / (float)img_h_p[0];
    bool ok = (cz > eps) && (un > 0.f) && (un < 1.f) && (vn > 0.f) && (vn < 1.f);
    ref2d[(size_t)i * 2 + 0] = un;
    ref2d[(size_t)i * 2 + 1] = vn;
    mask[i] = ok ? 1.f : 0.f;
}

// ---------------------------------------------------------------------------
// Kernel 2: fused offsets+attn-logits GEMM: (q+qpos)[NQ x 256] @ [Wo|Wa][256 x 384]
// Block: 256 threads = 8 waves (4M x 2N); block tile 128x128; wave tile 32x64.
// ---------------------------------------------------------------------------
__global__ __launch_bounds__(256) void qproj_kernel(
    const float* __restrict__ query, const float* __restrict__ qpos,
    const float* __restrict__ Wo, const float* __restrict__ bo,
    const float* __restrict__ Wa, const float* __restrict__ ba,
    float* __restrict__ off, float* __restrict__ attnw)
{
    __shared__ _Float16 As[128 * LDSA_STRIDE];
    __shared__ _Float16 BsT[128 * LDSB_STRIDE];
    int tid  = threadIdx.x;
    int wave = tid >> 5;
    int wm   = (wave >> 1) * 32;      // 0,32,64,96
    int wn   = (wave & 1) * 64;       // 0,64
    int row0 = blockIdx.x * 128;
    int col0 = blockIdx.y * 128;

    v8f zero = {0,0,0,0,0,0,0,0};
    v8f acc[2][4];
#pragma unroll
    for (int i = 0; i < 2; ++i)
#pragma unroll
        for (int j = 0; j < 4; ++j) acc[i][j] = zero;

    for (int k0 = 0; k0 < 256; k0 += 32) {
        // A: 128x32, coalesced along K
#pragma unroll
        for (int e = 0; e < 16; ++e) {
            int idx = tid + e * 256;
            int m = idx >> 5, k = idx & 31;
            int gr = row0 + m; if (gr > NQ - 1) gr = NQ - 1;
            size_t gi = (size_t)gr * 256 + (size_t)(k0 + k);
            if (e == 0) __builtin_prefetch(&query[gi + 32], 0, 1);  // speculative
            float val = query[gi] + qpos[gi];
            As[m * LDSA_STRIDE + k] = (_Float16)val;
        }
        // B: 32x128, coalesced along N, stored transposed
#pragma unroll
        for (int e = 0; e < 16; ++e) {
            int idx = tid + e * 256;
            int n = idx & 127, k = idx >> 7;
            int gn = col0 + n;
            float val = (gn < 256) ? Wo[(size_t)(k0 + k) * 256 + gn]
                                   : Wa[(size_t)(k0 + k) * 128 + (gn - 256)];
            BsT[n * LDSB_STRIDE + k] = (_Float16)val;
        }
        __syncthreads();
        v16h a0 = load_frag_a(As, wm);
        v16h a1 = load_frag_a(As, wm + 16);
        v16h b0 = load_frag_b(BsT, wn);
        v16h b1 = load_frag_b(BsT, wn + 16);
        v16h b2 = load_frag_b(BsT, wn + 32);
        v16h b3 = load_frag_b(BsT, wn + 48);
        acc[0][0] = WMMA_F16(a0, b0, acc[0][0]);
        acc[0][1] = WMMA_F16(a0, b1, acc[0][1]);
        acc[0][2] = WMMA_F16(a0, b2, acc[0][2]);
        acc[0][3] = WMMA_F16(a0, b3, acc[0][3]);
        acc[1][0] = WMMA_F16(a1, b0, acc[1][0]);
        acc[1][1] = WMMA_F16(a1, b1, acc[1][1]);
        acc[1][2] = WMMA_F16(a1, b2, acc[1][2]);
        acc[1][3] = WMMA_F16(a1, b3, acc[1][3]);
        __syncthreads();
    }

    int lane = tid & 31;
    int hi = lane >> 4;
#pragma unroll
    for (int ti = 0; ti < 2; ++ti) {
#pragma unroll
        for (int tj = 0; tj < 4; ++tj) {
            int m0 = row0 + wm + ti * 16;
            int n  = col0 + wn + tj * 16 + (lane & 15);
            float bias = (n < 256) ? bo[n] : ba[n - 256];
#pragma unroll
            for (int r = 0; r < 8; ++r) {
                int m = m0 + r + 8 * hi;
                if (m < NQ) {
                    float val = acc[ti][tj][r] + bias;
                    if (n < 256) off[(size_t)m * 256 + n] = val;
                    else         attnw[(size_t)m * 128 + (n - 256)] = val;
                }
            }
        }
    }
}

// ---------------------------------------------------------------------------
// Kernel 3: softmax over 16 (LEVELS*POINTS) per (q, head), in place
// ---------------------------------------------------------------------------
__global__ __launch_bounds__(256) void softmax16_kernel(float* __restrict__ attnw)
{
    int i = blockIdx.x * blockDim.x + threadIdx.x;   // q*HEADS + h
    if (i >= NQ * HEADS) return;
    float* p = attnw + (size_t)i * 16;
    float mx = -3.0e38f;
#pragma unroll
    for (int j = 0; j < 16; ++j) mx = fmaxf(mx, p[j]);
    float e[16], s = 0.f;
#pragma unroll
    for (int j = 0; j < 16; ++j) { e[j] = __expf(p[j] - mx); s += e[j]; }
    float inv = 1.f / s;
#pragma unroll
    for (int j = 0; j < 16; ++j) p[j] = e[j] * inv;
}

// ---------------------------------------------------------------------------
// Kernel 4: value projection per cam: v = feat^T @ Wv + bv, output f16 in
// [cam][head][hw][32] layout. feat is (CAMS, 256, H, W) channel-major.
// Block tile 128(hw) x 128(n); wave tile 32x64.
// ---------------------------------------------------------------------------
__global__ __launch_bounds__(256) void vproj_kernel(
    const float* __restrict__ f0, const float* __restrict__ f1,
    const float* __restrict__ f2, const float* __restrict__ f3,
    const float* __restrict__ Wv, const float* __restrict__ bv,
    _Float16* __restrict__ v)
{
    __shared__ _Float16 As[128 * LDSA_STRIDE];
    __shared__ _Float16 BsT[128 * LDSB_STRIDE];
    int tid  = threadIdx.x;
    int wave = tid >> 5;
    int wm   = (wave >> 1) * 32;
    int wn   = (wave & 1) * 64;
    int row0 = blockIdx.x * 128;    // hw tile
    int col0 = blockIdx.y * 128;    // output-channel tile
    int cam  = blockIdx.z;

    v8f zero = {0,0,0,0,0,0,0,0};
    v8f acc[2][4];
#pragma unroll
    for (int i = 0; i < 2; ++i)
#pragma unroll
        for (int j = 0; j < 4; ++j) acc[i][j] = zero;

    for (int k0 = 0; k0 < 256; k0 += 32) {
        // A: feat^T tile 128(hw) x 32(c): coalesce along hw for fixed channel
#pragma unroll
        for (int e = 0; e < 16; ++e) {
            int idx = tid + e * 256;
            int m = idx & 127, k = idx >> 7;
            int gm = row0 + m; if (gm > TOTHW - 1) gm = TOTHW - 1;
            int lvl = (gm >= 23200) + (gm >= 29000) + (gm >= 30450);
            const float* f = (lvl == 0) ? f0 : (lvl == 1) ? f1 : (lvl == 2) ? f2 : f3;
            int hwl = (lvl == 0) ? 23200 : (lvl == 1) ? 5800 : (lvl == 2) ? 1450 : 375;
            int lo  = (lvl == 0) ? 0 : (lvl == 1) ? 23200 : (lvl == 2) ? 29000 : 30450;
            const float* src = &f[((size_t)cam * 256 + (size_t)(k0 + k)) * (size_t)hwl + (size_t)(gm - lo)];
            if (e == 0) __builtin_prefetch(src + (size_t)hwl * 32, 0, 1);  // next K tile, speculative
            As[m * LDSA_STRIDE + k] = (_Float16)(*src);
        }
        // B: Wv tile 32x128, transposed into LDS
#pragma unroll
        for (int e = 0; e < 16; ++e) {
            int idx = tid + e * 256;
            int n = idx & 127, k = idx >> 7;
            BsT[n * LDSB_STRIDE + k] = (_Float16)Wv[(size_t)(k0 + k) * 256 + (col0 + n)];
        }
        __syncthreads();
        v16h a0 = load_frag_a(As, wm);
        v16h a1 = load_frag_a(As, wm + 16);
        v16h b0 = load_frag_b(BsT, wn);
        v16h b1 = load_frag_b(BsT, wn + 16);
        v16h b2 = load_frag_b(BsT, wn + 32);
        v16h b3 = load_frag_b(BsT, wn + 48);
        acc[0][0] = WMMA_F16(a0, b0, acc[0][0]);
        acc[0][1] = WMMA_F16(a0, b1, acc[0][1]);
        acc[0][2] = WMMA_F16(a0, b2, acc[0][2]);
        acc[0][3] = WMMA_F16(a0, b3, acc[0][3]);
        acc[1][0] = WMMA_F16(a1, b0, acc[1][0]);
        acc[1][1] = WMMA_F16(a1, b1, acc[1][1]);
        acc[1][2] = WMMA_F16(a1, b2, acc[1][2]);
        acc[1][3] = WMMA_F16(a1, b3, acc[1][3]);
        __syncthreads();
    }

    int lane = tid & 31;
    int hi = lane >> 4;
#pragma unroll
    for (int ti = 0; ti < 2; ++ti) {
#pragma unroll
        for (int tj = 0; tj < 4; ++tj) {
            int m0 = row0 + wm + ti * 16;
            int n  = col0 + wn + tj * 16 + (lane & 15);
            int head = n >> 5, d = n & 31;
            float bias = bv[n];
#pragma unroll
            for (int r = 0; r < 8; ++r) {
                int m = m0 + r + 8 * hi;
                if (m < TOTHW) {
                    float val = acc[ti][tj][r] + bias;
                    v[(((size_t)(cam * HEADS + head)) * TOTHW + (size_t)m) * 32 + d] = (_Float16)val;
                }
            }
        }
    }
}

// ---------------------------------------------------------------------------
// Kernel 5: deformable sampling + per-cam masked accumulation.
// Block = one query (256 threads); wave = one head; lane = head dim.
// ---------------------------------------------------------------------------
__global__ __launch_bounds__(256) void sample_kernel(
    const _Float16* __restrict__ v, const float* __restrict__ off,
    const float* __restrict__ attnw, const float* __restrict__ ref2d,
    const float* __restrict__ mask, float* __restrict__ agg,
    float* __restrict__ valid)
{
    int q = blockIdx.x;
    int h = threadIdx.x >> 5;
    int d = threadIdx.x & 31;

    const int Hs[4]   = {116, 58, 29, 15};
    const int Ws[4]   = {200, 100, 50, 25};
    const int Offs[4] = {0, 23200, 29000, 30450};

    float acc = 0.f;
    float vcount = 0.f;

    for (int cam = 0; cam < CAMS; ++cam) {
        float m = mask[cam * NQ + q];
        vcount += m;
        if (m == 0.f) continue;                    // wave-uniform branch
        float rx = ref2d[((size_t)cam * NQ + q) * 2 + 0];
        float ry = ref2d[((size_t)cam * NQ + q) * 2 + 1];
        const _Float16* vcamh = v + ((size_t)(cam * HEADS + h)) * TOTHW * 32;
#pragma unroll
        for (int l = 0; l < LEVELS; ++l) {
            const int Hl = Hs[l], Wl = Ws[l];
            const _Float16* vb = vcamh + (size_t)Offs[l] * 32;
            float fW = (float)Wl, fH = (float)Hl;
#pragma unroll
            for (int p = 0; p < POINTS; ++p) {
                int oidx = q * 256 + h * 64 + l * 16 + p * 2;
                float lx = rx + off[oidx]     / fW;
                float ly = ry + off[oidx + 1] / fH;
                float aw = attnw[q * 128 + h * 16 + l * 4 + p];
                float x = lx * fW - 0.5f;
                float y = ly * fH - 0.5f;
                float xf = floorf(x), yf = floorf(y);
                float wx = x - xf, wy = y - yf;
                int x0 = (int)xf, y0 = (int)yf;
                auto corner = [&](int xi, int yi) -> float {
                    if (xi < 0 || xi >= Wl || yi < 0 || yi >= Hl) return 0.f;
                    return (float)vb[((size_t)(yi * Wl + xi)) * 32 + d];
                };
                float s = corner(x0, y0)         * (1.f - wx) * (1.f - wy)
                        + corner(x0 + 1, y0)     * wx         * (1.f - wy)
                        + corner(x0, y0 + 1)     * (1.f - wx) * wy
                        + corner(x0 + 1, y0 + 1) * wx         * wy;
                acc += aw * s;
            }
        }
    }
    agg[(size_t)q * 256 + h * 32 + d] = acc;
    if (threadIdx.x == 0) valid[q] = vcount;
}

// ---------------------------------------------------------------------------
// Kernel 6: output projection: out = (agg @ Wout + bout*valid) / max(valid,1)
// Block tile 128x128; wave tile 32x64.
// ---------------------------------------------------------------------------
__global__ __launch_bounds__(256) void outproj_kernel(
    const float* __restrict__ agg, const float* __restrict__ Wout,
    const float* __restrict__ bout, const float* __restrict__ valid,
    float* __restrict__ out)
{
    __shared__ _Float16 As[128 * LDSA_STRIDE];
    __shared__ _Float16 BsT[128 * LDSB_STRIDE];
    int tid  = threadIdx.x;
    int wave = tid >> 5;
    int wm   = (wave >> 1) * 32;
    int wn   = (wave & 1) * 64;
    int row0 = blockIdx.x * 128;
    int col0 = blockIdx.y * 128;

    v8f zero = {0,0,0,0,0,0,0,0};
    v8f acc[2][4];
#pragma unroll
    for (int i = 0; i < 2; ++i)
#pragma unroll
        for (int j = 0; j < 4; ++j) acc[i][j] = zero;

    for (int k0 = 0; k0 < 256; k0 += 32) {
#pragma unroll
        for (int e = 0; e < 16; ++e) {
            int idx = tid + e * 256;
            int m = idx >> 5, k = idx & 31;
            int gr = row0 + m; if (gr > NQ - 1) gr = NQ - 1;
            size_t gi = (size_t)gr * 256 + (size_t)(k0 + k);
            if (e == 0) __builtin_prefetch(&agg[gi + 32], 0, 1);   // speculative
            As[m * LDSA_STRIDE + k] = (_Float16)agg[gi];
        }
#pragma unroll
        for (int e = 0; e < 16; ++e) {
            int idx = tid + e * 256;
            int n = idx & 127, k = idx >> 7;
            BsT[n * LDSB_STRIDE + k] = (_Float16)Wout[(size_t)(k0 + k) * 256 + (col0 + n)];
        }
        __syncthreads();
        v16h a0 = load_frag_a(As, wm);
        v16h a1 = load_frag_a(As, wm + 16);
        v16h b0 = load_frag_b(BsT, wn);
        v16h b1 = load_frag_b(BsT, wn + 16);
        v16h b2 = load_frag_b(BsT, wn + 32);
        v16h b3 = load_frag_b(BsT, wn + 48);
        acc[0][0] = WMMA_F16(a0, b0, acc[0][0]);
        acc[0][1] = WMMA_F16(a0, b1, acc[0][1]);
        acc[0][2] = WMMA_F16(a0, b2, acc[0][2]);
        acc[0][3] = WMMA_F16(a0, b3, acc[0][3]);
        acc[1][0] = WMMA_F16(a1, b0, acc[1][0]);
        acc[1][1] = WMMA_F16(a1, b1, acc[1][1]);
        acc[1][2] = WMMA_F16(a1, b2, acc[1][2]);
        acc[1][3] = WMMA_F16(a1, b3, acc[1][3]);
        __syncthreads();
    }

    int lane = tid & 31;
    int hi = lane >> 4;
#pragma unroll
    for (int ti = 0; ti < 2; ++ti) {
#pragma unroll
        for (int tj = 0; tj < 4; ++tj) {
            int m0 = row0 + wm + ti * 16;
            int n  = col0 + wn + tj * 16 + (lane & 15);
            float bias = bout[n];
#pragma unroll
            for (int r = 0; r < 8; ++r) {
                int m = m0 + r + 8 * hi;
                if (m < NQ) {
                    float vd = valid[m];
                    float val = (acc[ti][tj][r] + bias * vd) / fmaxf(vd, 1.f);
                    out[(size_t)m * 256 + n] = val;
                }
            }
        }
    }
}

// ---------------------------------------------------------------------------
// Host launcher
// ---------------------------------------------------------------------------
extern "C" void kernel_launch(void* const* d_in, const int* in_sizes, int n_in,
                              void* d_out, int out_size, void* d_ws, size_t ws_size,
                              hipStream_t stream)
{
    const float* query  = (const float*)d_in[0];
    const float* qpos   = (const float*)d_in[1];
    const float* l2i    = (const float*)d_in[2];
    const float* refp   = (const float*)d_in[3];
    const float* f0     = (const float*)d_in[4];
    const float* f1     = (const float*)d_in[5];
    const float* f2     = (const float*)d_in[6];
    const float* f3     = (const float*)d_in[7];
    const float* Wv     = (const float*)d_in[8];
    const float* bv     = (const float*)d_in[9];
    const float* Wo     = (const float*)d_in[10];
    const float* bo     = (const float*)d_in[11];
    const float* Wa     = (const float*)d_in[12];
    const float* ba     = (const float*)d_in[13];
    const float* Wout   = (const float*)d_in[14];
    const float* bout   = (const float*)d_in[15];
    const int*   img_h  = (const int*)d_in[16];
    const int*   img_w  = (const int*)d_in[17];
    float* out = (float*)d_out;

    // Workspace layout (256B aligned)
    char* ws = (char*)d_ws;
    auto align256 = [](size_t x) { return (x + 255) & ~(size_t)255; };
    size_t o = 0;
    float*    ref2d = (float*)(ws + o);    o = align256(o + (size_t)CAMS * NQ * 2 * 4);
    float*    maskb = (float*)(ws + o);    o = align256(o + (size_t)CAMS * NQ * 4);
    float*    offb  = (float*)(ws + o);    o = align256(o + (size_t)NQ * 256 * 4);
    float*    attnw = (float*)(ws + o);    o = align256(o + (size_t)NQ * 128 * 4);
    _Float16* vbuf  = (_Float16*)(ws + o); o = align256(o + (size_t)CAMS * HEADS * TOTHW * 32 * 2);
    float*    agg   = (float*)(ws + o);    o = align256(o + (size_t)NQ * 256 * 4);
    float*    valid = (float*)(ws + o);    o = align256(o + (size_t)NQ * 4);
    (void)ws_size; (void)in_sizes; (void)n_in; (void)out_size;

    // 1. point projection
    point_kernel<<<dim3((CAMS * NQ + 255) / 256), dim3(256), 0, stream>>>(
        l2i, refp, img_h, img_w, ref2d, maskb);

    // 2. offsets + attention logits (camera-invariant, computed once)
    qproj_kernel<<<dim3((NQ + 127) / 128, 3), dim3(256), 0, stream>>>(
        query, qpos, Wo, bo, Wa, ba, offb, attnw);

    // 3. softmax over 16
    softmax16_kernel<<<dim3((NQ * HEADS + 255) / 256), dim3(256), 0, stream>>>(attnw);

    // 4. value projection per camera (f16 output, head-major layout)
    vproj_kernel<<<dim3((TOTHW + 127) / 128, 2, CAMS), dim3(256), 0, stream>>>(
        f0, f1, f2, f3, Wv, bv, vbuf);

    // 5. deformable sampling + masked accumulation over cameras
    sample_kernel<<<dim3(NQ), dim3(256), 0, stream>>>(
        vbuf, offb, attnw, ref2d, maskb, agg, valid);

    // 6. single output projection with mask normalization
    outproj_kernel<<<dim3((NQ + 127) / 128, 2), dim3(256), 0, stream>>>(
        agg, Wout, bout, valid, out);
}
